// FlowModelTrunk_10445360464051
// MI455X (gfx1250) — compile-verified, matrix operand order checked by reading, not run on hardware
//
#include <hip/hip_runtime.h>
#include <hip/hip_bf16.h>

// =====================================================================
// FlowModelTrunk forward for MI455X (gfx1250, wave32, WMMA).
// All matmuls use v_wmma_f32_16x16x32_f16 (f16 in, f32 acc).
//  - gemm_wmma_fast: branch-free 2x2 macro-tile (32x32 per wave),
//    4 WMMAs per K-step from 4 b128 fragment loads (A + B-NK layout is
//    two contiguous 16B runs per lane). Used for every GEMM with
//    M%32==0, N%32==0, K%32==0 (all the heavy ones).
//  - gemm_wmma_edge: guarded single-tile path for ragged shapes
//    (N=6/12/16/24/144/432, K=16) -- <1% of total FLOPs.
// Softmax / layernorm / rigid-frame math stay in f32 VALU with wave32
// shuffle reductions.
// =====================================================================

typedef _Float16 f16;
typedef _Float16 v16h __attribute__((ext_vector_type(16)));
typedef _Float16 v8h  __attribute__((ext_vector_type(8)));
typedef float    v8f  __attribute__((ext_vector_type(8)));

#define WMMA16(av, bv, acc) __builtin_amdgcn_wmma_f32_16x16x32_f16( \
        false, av, false, bv, (short)0, acc, false, false)

// ---------------------------------------------------------------------
// FAST: C[b] = act(alpha * A[b] @ op(B[b]) + bias) (+resid)
// Requires M%32==0, N%32==0, K%32==0; all rows/cols/batch offsets
// 8-element (16B) aligned. One wave = 32x32 output macro-tile.
// ---------------------------------------------------------------------
template<bool B_IS_NK, bool OUT_F16, bool RELU, bool RESID>
__global__ __launch_bounds__(256)
void gemm_wmma_fast(const f16* __restrict__ A, const f16* __restrict__ B,
                    void* __restrict__ C, const float* __restrict__ bias,
                    const f16* __restrict__ resid,
                    int M, int N, int K, int lda, int ldb, int ldc,
                    long long sA, long long sB, long long sC,
                    int batch, float alpha)
{
    int tilesM = M >> 5;
    int tilesN = N >> 5;
    long long tilesPer = (long long)tilesM * tilesN;
    long long tile = (long long)blockIdx.x * (blockDim.x >> 5) + (threadIdx.x >> 5);
    if (tile >= tilesPer * batch) return;

    int b   = (int)(tile / tilesPer);
    int rem = (int)(tile - (long long)b * tilesPer);
    int tm  = rem / tilesN;
    int tn  = rem % tilesN;

    const f16* Ab = A + (long long)b * sA;
    const f16* Bb = B + (long long)b * sB;

    int lane = threadIdx.x & 31;
    int hlf  = lane >> 4;
    int l16  = lane & 15;
    int m0 = tm * 32, n0 = tn * 32;

    const f16* ap0 = Ab + (long long)(m0 + l16) * lda + hlf * 8;
    const f16* ap1 = ap0 + (long long)16 * lda;
    const f16* bp0;
    const f16* bp1;
    if (B_IS_NK) {
        bp0 = Bb + (long long)(n0 + l16) * ldb + hlf * 8;
        bp1 = bp0 + (long long)16 * ldb;
    } else {
        bp0 = Bb + (n0 + l16);
        bp1 = bp0 + 16;
    }

    v8f acc00 = {}, acc01 = {}, acc10 = {}, acc11 = {};
    for (int kk = 0; kk < K; kk += 32) {
        v16h av0, av1, bv0, bv1;
        {
            v8h x0 = *(const v8h*)(ap0 + kk);
            v8h x1 = *(const v8h*)(ap0 + kk + 16);
            v8h y0 = *(const v8h*)(ap1 + kk);
            v8h y1 = *(const v8h*)(ap1 + kk + 16);
            #pragma unroll
            for (int i = 0; i < 8; ++i) {
                av0[i] = x0[i]; av0[8 + i] = x1[i];
                av1[i] = y0[i]; av1[8 + i] = y1[i];
            }
        }
        if (B_IS_NK) {
            v8h x0 = *(const v8h*)(bp0 + kk);
            v8h x1 = *(const v8h*)(bp0 + kk + 16);
            v8h y0 = *(const v8h*)(bp1 + kk);
            v8h y1 = *(const v8h*)(bp1 + kk + 16);
            #pragma unroll
            for (int i = 0; i < 8; ++i) {
                bv0[i] = x0[i]; bv0[8 + i] = x1[i];
                bv1[i] = y0[i]; bv1[8 + i] = y1[i];
            }
        } else {
            #pragma unroll
            for (int i = 0; i < 8; ++i) {
                int k = kk + hlf * 8 + (i >> 2) * 16 + (i & 3) * 2;
                const f16* r0 = bp0 + (long long)k * ldb;
                const f16* r1 = r0 + ldb;
                bv0[2 * i]     = r0[0];
                bv0[2 * i + 1] = r1[0];
                bv1[2 * i]     = r0[16];
                bv1[2 * i + 1] = r1[16];
            }
        }
        acc00 = WMMA16(av0, bv0, acc00);
        acc01 = WMMA16(av0, bv1, acc01);
        acc10 = WMMA16(av1, bv0, acc10);
        acc11 = WMMA16(av1, bv1, acc11);
    }

    int cn0 = n0 + l16, cn1 = cn0 + 16;
    float bi0 = bias ? bias[cn0] : 0.f;
    float bi1 = bias ? bias[cn1] : 0.f;
    #pragma unroll
    for (int r = 0; r < 8; ++r) {
        int cm0 = m0 + r + hlf * 8;
        int cm1 = cm0 + 16;
        float v00 = acc00[r] * alpha + bi0;
        float v01 = acc01[r] * alpha + bi1;
        float v10 = acc10[r] * alpha + bi0;
        float v11 = acc11[r] * alpha + bi1;
        if (RELU) {
            v00 = v00 > 0.f ? v00 : 0.f;  v01 = v01 > 0.f ? v01 : 0.f;
            v10 = v10 > 0.f ? v10 : 0.f;  v11 = v11 > 0.f ? v11 : 0.f;
        }
        long long o00 = (long long)b * sC + (long long)cm0 * ldc + cn0;
        long long o01 = (long long)b * sC + (long long)cm0 * ldc + cn1;
        long long o10 = (long long)b * sC + (long long)cm1 * ldc + cn0;
        long long o11 = (long long)b * sC + (long long)cm1 * ldc + cn1;
        if (RESID) {
            v00 += (float)resid[o00]; v01 += (float)resid[o01];
            v10 += (float)resid[o10]; v11 += (float)resid[o11];
        }
        if (OUT_F16) {
            ((f16*)C)[o00] = (f16)v00; ((f16*)C)[o01] = (f16)v01;
            ((f16*)C)[o10] = (f16)v10; ((f16*)C)[o11] = (f16)v11;
        } else {
            ((float*)C)[o00] = v00; ((float*)C)[o01] = v01;
            ((float*)C)[o10] = v10; ((float*)C)[o11] = v11;
        }
    }
}

// ---------------------------------------------------------------------
// EDGE: guarded single-tile path for ragged shapes. Zero-pads at edges.
// ---------------------------------------------------------------------
template<bool B_IS_NK, bool OUT_F16>
__global__ __launch_bounds__(256)
void gemm_wmma_edge(const f16* __restrict__ A, const f16* __restrict__ B,
                    void* __restrict__ C, const float* __restrict__ bias,
                    int M, int N, int K, int lda, int ldb, int ldc,
                    long long sA, long long sB, long long sC,
                    int batch, float alpha)
{
    int tilesM = (M + 15) >> 4;
    int tilesN = (N + 15) >> 4;
    long long tilesPer = (long long)tilesM * tilesN;
    long long tile = (long long)blockIdx.x * (blockDim.x >> 5) + (threadIdx.x >> 5);
    if (tile >= tilesPer * batch) return;

    int b   = (int)(tile / tilesPer);
    int rem = (int)(tile - (long long)b * tilesPer);
    int tm  = rem / tilesN;
    int tn  = rem % tilesN;

    const f16* Ab = A + (long long)b * sA;
    const f16* Bb = B + (long long)b * sB;

    int lane = threadIdx.x & 31;
    int hlf  = lane >> 4;
    int l16  = lane & 15;
    int m0 = tm * 16, n0 = tn * 16;

    v8f acc = {};
    for (int kk = 0; kk < K; kk += 32) {
        v16h av, bv;
        int arow = m0 + l16;
        #pragma unroll
        for (int i = 0; i < 8; ++i) {
            int k = kk + hlf * 8 + (i >> 2) * 16 + (i & 3) * 2;
            f16 e0 = (f16)0.f, e1 = (f16)0.f;
            if (arow < M) {
                long long ro = (long long)arow * lda;
                if (k     < K) e0 = Ab[ro + k];
                if (k + 1 < K) e1 = Ab[ro + k + 1];
            }
            av[2 * i] = e0; av[2 * i + 1] = e1;
        }
        int bcol = n0 + l16;
        #pragma unroll
        for (int i = 0; i < 8; ++i) {
            int k = kk + hlf * 8 + (i >> 2) * 16 + (i & 3) * 2;
            f16 e0 = (f16)0.f, e1 = (f16)0.f;
            if (bcol < N) {
                if (B_IS_NK) {
                    long long ro = (long long)bcol * ldb;
                    if (k     < K) e0 = Bb[ro + k];
                    if (k + 1 < K) e1 = Bb[ro + k + 1];
                } else {
                    if (k     < K) e0 = Bb[(long long)k * ldb + bcol];
                    if (k + 1 < K) e1 = Bb[(long long)(k + 1) * ldb + bcol];
                }
            }
            bv[2 * i] = e0; bv[2 * i + 1] = e1;
        }
        acc = WMMA16(av, bv, acc);
    }

    int cn = n0 + l16;
    #pragma unroll
    for (int r = 0; r < 8; ++r) {
        int cm = m0 + r + hlf * 8;
        if (cm < M && cn < N) {
            float v = acc[r] * alpha;
            if (bias) v += bias[cn];
            long long off = (long long)b * sC + (long long)cm * ldc + cn;
            if (OUT_F16) ((f16*)C)[off] = (f16)v;
            else         ((float*)C)[off] = v;
        }
    }
}

template<bool BNK, bool OF16, bool RELU, bool RES>
static void run_fast(hipStream_t st, const f16* A, const f16* B, void* C,
                     const float* bias, const f16* resid,
                     int M, int N, int K, int lda, int ldb, int ldc,
                     long long sA, long long sB, long long sC,
                     int batch, float alpha)
{
    long long tiles = (long long)(M / 32) * (N / 32) * batch;
    long long blocks = (tiles + 7) / 8;                     // 8 waves / block
    gemm_wmma_fast<BNK, OF16, RELU, RES><<<dim3((unsigned)blocks), dim3(256), 0, st>>>(
        A, B, C, bias, resid, M, N, K, lda, ldb, ldc, sA, sB, sC, batch, alpha);
}

template<bool BNK, bool OF16>
static void run_edge(hipStream_t st, const f16* A, const f16* B, void* C,
                     const float* bias,
                     int M, int N, int K, int lda, int ldb, int ldc,
                     long long sA, long long sB, long long sC,
                     int batch, float alpha)
{
    long long tiles = (long long)((M + 15) / 16) * ((N + 15) / 16) * batch;
    long long blocks = (tiles + 7) / 8;
    gemm_wmma_edge<BNK, OF16><<<dim3((unsigned)blocks), dim3(256), 0, st>>>(
        A, B, C, bias, M, N, K, lda, ldb, ldc, sA, sB, sC, batch, alpha);
}

// ---------------------------------------------------------------------
// Elementwise / reduction kernels
// ---------------------------------------------------------------------
__global__ void k_cvt(const float* __restrict__ in, f16* __restrict__ out, long long n)
{
    long long id = (long long)blockIdx.x * blockDim.x + threadIdx.x;
    if (id < n) out[id] = (f16)in[id];
}

// Rigid-frame point construction: pts[i][p][3] = R_i @ x + t_i.
// lin layout: [i][c*npts + p] (coordinate-major, as in the reference reshape).
// For the kv variant also emit f16 v_pts laid out [j][h*24 + (p-4)*3 + c].
__global__ void k_make_pts(const float* __restrict__ lin, const float* __restrict__ rot,
                           const float* __restrict__ trans, float* __restrict__ pts,
                           f16* __restrict__ v16, int npts, int Phead)
{
    int id = blockIdx.x * blockDim.x + threadIdx.x;
    if (id >= 256 * npts) return;
    int i = id / npts, p = id % npts;
    const float* L = lin + (long long)i * 3 * npts;
    float x = L[p], y = L[npts + p], z = L[2 * npts + p];
    const float* R = rot + i * 9;
    float o0 = R[0] * x + R[1] * y + R[2] * z + trans[i * 3 + 0];
    float o1 = R[3] * x + R[4] * y + R[5] * z + trans[i * 3 + 1];
    float o2 = R[6] * x + R[7] * y + R[8] * z + trans[i * 3 + 2];
    long long po = (long long)i * npts * 3 + p * 3;
    pts[po] = o0; pts[po + 1] = o1; pts[po + 2] = o2;
    if (v16) {
        int h = p / Phead, pp = p % Phead;
        if (pp >= 4) {
            long long vo = (long long)i * 288 + h * 24 + (pp - 4) * 3;
            v16[vo] = (f16)o0; v16[vo + 1] = (f16)o1; v16[vo + 2] = (f16)o2;
        }
    }
}

// IPA logits combine + softmax. One wave per (h,i) row of length 256.
// logit = qk(scaled) + sqrt(1/3)*b_bias[i,j,h] - 0.5*hw_h*||qp-kp||^2 + mask
__global__ void k_ipa_softmax(const float* __restrict__ qk, const float* __restrict__ bbias,
                              const float* __restrict__ qpts, const float* __restrict__ kvpts,
                              const float* __restrict__ mask, const float* __restrict__ headw,
                              f16* __restrict__ a16)
{
    int wave = (blockIdx.x * blockDim.x + threadIdx.x) >> 5;
    int lane = threadIdx.x & 31;
    if (wave >= 12 * 256) return;
    int h = wave >> 8, i = wave & 255;
    float hw = logf(1.f + expf(headw[h])) * 0.13608276348795434f; // softplus * sqrt(1/54)
    float qd[12];
    #pragma unroll
    for (int t = 0; t < 12; ++t) qd[t] = qpts[(long long)i * 144 + h * 12 + t];
    float mi = mask[i];
    const float SB = 0.57735026918962576f; // sqrt(1/3)
    float lv[8];
    float mx = -3.4e38f;
    for (int u = 0; u < 8; ++u) {
        int j = lane + u * 32;
        float pt = 0.f;
        #pragma unroll
        for (int p = 0; p < 4; ++p) {
            long long ko = (long long)j * 432 + h * 36 + p * 3;
            float dx = qd[p * 3 + 0] - kvpts[ko];
            float dy = qd[p * 3 + 1] - kvpts[ko + 1];
            float dz = qd[p * 3 + 2] - kvpts[ko + 2];
            pt += dx * dx + dy * dy + dz * dz;
        }
        float l = qk[((long long)h * 256 + i) * 256 + j]
                + SB * bbias[((long long)i * 256 + j) * 12 + h]
                - 0.5f * hw * pt
                + 100000.0f * (mi * mask[j] - 1.f);
        lv[u] = l;
        mx = fmaxf(mx, l);
    }
    for (int o = 16; o; o >>= 1) mx = fmaxf(mx, __shfl_xor(mx, o, 32));
    float s = 0.f;
    for (int u = 0; u < 8; ++u) { lv[u] = expf(lv[u] - mx); s += lv[u]; }
    for (int o = 16; o; o >>= 1) s += __shfl_xor(s, o, 32);
    float inv = 1.f / s;
    for (int u = 0; u < 8; ++u)
        a16[((long long)h * 256 + i) * 256 + lane + u * 32] = (f16)(lv[u] * inv);
}

// Transformer softmax: one wave per (h,i) row (4 heads), key-mask added.
__global__ void k_tf_softmax(const float* __restrict__ att, const float* __restrict__ mask,
                             f16* __restrict__ a16)
{
    int wave = (blockIdx.x * blockDim.x + threadIdx.x) >> 5;
    int lane = threadIdx.x & 31;
    if (wave >= 4 * 256) return;
    float lv[8];
    float mx = -3.4e38f;
    for (int u = 0; u < 8; ++u) {
        int j = lane + u * 32;
        float l = att[(long long)wave * 256 + j] + (mask[j] > 0.f ? 0.f : -1e9f);
        lv[u] = l; mx = fmaxf(mx, l);
    }
    for (int o = 16; o; o >>= 1) mx = fmaxf(mx, __shfl_xor(mx, o, 32));
    float s = 0.f;
    for (int u = 0; u < 8; ++u) { lv[u] = expf(lv[u] - mx); s += lv[u]; }
    for (int o = 16; o; o >>= 1) s += __shfl_xor(s, o, 32);
    float inv = 1.f / s;
    for (int u = 0; u < 8; ++u)
        a16[(long long)wave * 256 + lane + u * 32] = (f16)(lv[u] * inv);
}

// LayerNorm(x + resid*maskfac)*g + b. One wave per row, C in {128,384}.
// out32 / out16 optional; supports in-place (each lane reads its own
// elements into registers before any write).
__global__ void k_ln(const float* __restrict__ x, const float* __restrict__ resid,
                     const float* __restrict__ mask, const float* __restrict__ g,
                     const float* __restrict__ bb, float* __restrict__ out32,
                     f16* __restrict__ out16, int rows, int C)
{
    int wave = (blockIdx.x * blockDim.x + threadIdx.x) >> 5;
    int lane = threadIdx.x & 31;
    if (wave >= rows) return;
    int per = C >> 5;
    float v[16];
    float mfac = mask ? mask[wave] : 1.f;
    float sum = 0.f;
    for (int t = 0; t < per; ++t) {
        long long off = (long long)wave * C + lane + t * 32;
        float xv = x[off];
        if (resid) xv += resid[off] * mfac;
        v[t] = xv; sum += xv;
    }
    for (int o = 16; o; o >>= 1) sum += __shfl_xor(sum, o, 32);
    float mu = sum / C;
    float var = 0.f;
    for (int t = 0; t < per; ++t) { float d = v[t] - mu; var += d * d; }
    for (int o = 16; o; o >>= 1) var += __shfl_xor(var, o, 32);
    float rs = rsqrtf(var / C + 1e-5f);
    for (int t = 0; t < per; ++t) {
        int c = lane + t * 32;
        float y = (v[t] - mu) * rs * g[c] + bb[c];
        long long off = (long long)wave * C + c;
        if (out32) out32[off] = y;
        if (out16) out16[off] = (f16)y;
    }
}

// Finish o_pt: local = R^T (o_pt - t); write x/y/z/norm blocks into feats.
__global__ void k_opt_finish(const float* __restrict__ optraw, const float* __restrict__ rot,
                             const float* __restrict__ trans, f16* __restrict__ feats)
{
    int id = blockIdx.x * blockDim.x + threadIdx.x;
    if (id >= 256 * 96) return;
    int i = id / 96, hp = id % 96;
    int h = hp >> 3, p = hp & 7;
    long long base = (long long)i * 288 + h * 24 + p * 3;
    float gx = optraw[base + 0] - trans[i * 3 + 0];
    float gy = optraw[base + 1] - trans[i * 3 + 1];
    float gz = optraw[base + 2] - trans[i * 3 + 2];
    const float* R = rot + i * 9;
    float lx = R[0] * gx + R[3] * gy + R[6] * gz;
    float ly = R[1] * gx + R[4] * gy + R[7] * gz;
    float lz = R[2] * gx + R[5] * gy + R[8] * gz;
    float nrm = sqrtf(lx * lx + ly * ly + lz * lz + 1e-5f);
    long long fb = (long long)i * 960;
    feats[fb + 192 + hp] = (f16)lx;
    feats[fb + 288 + hp] = (f16)ly;
    feats[fb + 384 + hp] = (f16)lz;
    feats[fb + 480 + hp] = (f16)nrm;
}

// Build edge concat features: e[i,j] = [z(128) | n_i(192) | n_j(192)]
__global__ void k_ecat(const f16* __restrict__ z16, const f16* __restrict__ n16,
                       f16* __restrict__ e)
{
    long long id = (long long)blockIdx.x * blockDim.x + threadIdx.x;
    if (id >= (long long)65536 * 512) return;
    long long row = id >> 9; int c = (int)(id & 511);
    int i = (int)(row >> 8), j = (int)(row & 255);
    f16 v;
    if (c < 128)      v = z16[row * 128 + c];
    else if (c < 320) v = n16[(long long)i * 192 + (c - 128)];
    else              v = n16[(long long)j * 192 + (c - 320)];
    e[id] = v;
}

// Final s = s * mask -> d_out (f32) and f16 copy for downstream GEMMs.
__global__ void k_mask_s(const float* __restrict__ s, const float* __restrict__ mask,
                         float* __restrict__ o32, f16* __restrict__ o16)
{
    int id = blockIdx.x * blockDim.x + threadIdx.x;
    if (id >= 256 * 384) return;
    float v = s[id] * mask[id / 384];
    o32[id] = v; o16[id] = (f16)v;
}

// ---------------------------------------------------------------------
// Host orchestration
// ---------------------------------------------------------------------
struct Bump {
    char* base; size_t off;
    template<class T> T* get(size_t n) {
        off = (off + 255) & ~(size_t)255;
        T* p = (T*)(base + off);
        off += n * sizeof(T);
        return p;
    }
};

static void cvt(hipStream_t st, const float* src, f16* dst, long long n)
{
    k_cvt<<<dim3((unsigned)((n + 255) / 256)), dim3(256), 0, st>>>(src, dst, n);
}

// d_in index map (setup_inputs dict insertion order, params flattened
// recursively in insertion order):
enum {
    IN_S = 0, IN_Z, IN_ROT, IN_TRANS, IN_MASK,
    I_WQ = 5, I_BQ, I_WKV, I_BKV, I_WQP, I_BQP, I_WKVP, I_BKVP,
    I_WB, I_BB, I_WDZ, I_BDZ, I_WOUT, I_BOUT, I_HW,
    I_LNG = 20, I_LNB = 21,
    T0 = 22, T1 = 34, // per layer: Wqkv,bqkv,Wo,bo,ln1g,ln1b,W1,b1,W2,b2,ln2g,ln2b
    S_W1 = 46, S_B1, S_W2, S_B2, S_W3, S_B3, S_LNG, S_LNB,
    I_BBW = 54, I_BBB = 55,
    E_WI = 56, E_BI, E_WT1, E_BT1, E_WT2, E_BT2, E_WF, E_BF, E_LNG, E_LNB
};

extern "C" void kernel_launch(void* const* d_in, const int* in_sizes, int n_in,
                              void* d_out, int out_size, void* d_ws, size_t ws_size,
                              hipStream_t stream)
{
    (void)in_sizes; (void)n_in; (void)out_size; (void)ws_size;
    const float* F  = nullptr;   // null bias
    const f16*   FR = nullptr;   // null residual
    auto in = [&](int i) { return (const float*)d_in[i]; };

    Bump ws{ (char*)d_ws, 0 };

    // ---- f16 workspace ----
    f16* x16    = ws.get<f16>(256 * 384);        // current s (f16), updated in place
    f16* z16    = ws.get<f16>(65536 * 128);
    f16* q16    = ws.get<f16>(256 * 192);
    f16* kv16   = ws.get<f16>(256 * 384);
    f16* vpts16 = ws.get<f16>(256 * 288);
    f16* a16    = ws.get<f16>(12 * 65536);       // IPA probs; reused for tfmr probs
    f16* pz16   = ws.get<f16>(65536 * 32);
    f16* feats  = ws.get<f16>(256 * 960);
    f16* qkv16  = ws.get<f16>(256 * 1152);
    f16* oatt16 = ws.get<f16>(256 * 384);
    f16* ffn16  = ws.get<f16>(256 * 384);
    f16* t1_16  = ws.get<f16>(256 * 384);
    f16* t2_16  = ws.get<f16>(256 * 384);
    f16* n16    = ws.get<f16>(256 * 192);
    f16* ecat   = ws.get<f16>((size_t)65536 * 512);  // e, later overwritten by h2+e
    f16* h1buf  = ws.get<f16>((size_t)65536 * 512);

    // ---- f16 weight copies ----
    f16* wq    = ws.get<f16>(192 * 384);
    f16* wkv   = ws.get<f16>(384 * 384);
    f16* wqp   = ws.get<f16>(144 * 384);
    f16* wkvp  = ws.get<f16>(432 * 384);
    f16* wb    = ws.get<f16>(12 * 128);
    f16* wdz   = ws.get<f16>(32 * 128);
    f16* wout  = ws.get<f16>(384 * 960);
    f16* wbbw  = ws.get<f16>(6 * 384);
    f16* wi    = ws.get<f16>(192 * 384);
    f16* wt1   = ws.get<f16>(512 * 512);
    f16* wt2   = ws.get<f16>(512 * 512);
    f16* wf    = ws.get<f16>(128 * 512);
    f16* ws1   = ws.get<f16>(384 * 384);
    f16* ws2   = ws.get<f16>(384 * 384);
    f16* ws3   = ws.get<f16>(384 * 384);
    f16* wqkvL[2], *woL[2], *w1L[2], *w2L[2];
    for (int l = 0; l < 2; ++l) {
        wqkvL[l] = ws.get<f16>(1152 * 384);
        woL[l]   = ws.get<f16>(384 * 384);
        w1L[l]   = ws.get<f16>(384 * 384);
        w2L[l]   = ws.get<f16>(384 * 384);
    }

    // ---- f32 workspace ----
    float* qp_lin  = ws.get<float>(256 * 144);
    float* kvp_lin = ws.get<float>(256 * 432);
    float* qpts    = ws.get<float>(256 * 144);
    float* kvpts   = ws.get<float>(256 * 432);
    float* bbias   = ws.get<float>((size_t)65536 * 12);
    float* qk32    = ws.get<float>((size_t)12 * 65536); // reused for tfmr logits
    float* optraw  = ws.get<float>(256 * 288);
    float* tmp32   = ws.get<float>(256 * 384);
    float* sbuf32  = ws.get<float>(256 * 384);

    float* outp    = (float*)d_out;
    float* out_s   = outp;                 // 256*384
    float* out_rig = outp + 256 * 384;     // 256*6
    float* out_e   = outp + 256 * 384 + 256 * 6; // 65536*128

    const float AQK = 0.14433756729740643f;  // sqrt(1/(3*16))
    const float ATF = 0.10206207261596575f;  // 1/sqrt(96)

    // ---- convert inputs & weights to f16 ----
    cvt(stream, in(IN_S), x16, 256 * 384);
    cvt(stream, in(IN_Z), z16, (long long)65536 * 128);
    cvt(stream, in(I_WQ), wq, 192 * 384);
    cvt(stream, in(I_WKV), wkv, 384 * 384);
    cvt(stream, in(I_WQP), wqp, 144 * 384);
    cvt(stream, in(I_WKVP), wkvp, 432 * 384);
    cvt(stream, in(I_WB), wb, 12 * 128);
    cvt(stream, in(I_WDZ), wdz, 32 * 128);
    cvt(stream, in(I_WOUT), wout, 384 * 960);
    cvt(stream, in(I_BBW), wbbw, 6 * 384);
    cvt(stream, in(E_WI), wi, 192 * 384);
    cvt(stream, in(E_WT1), wt1, 512 * 512);
    cvt(stream, in(E_WT2), wt2, 512 * 512);
    cvt(stream, in(E_WF), wf, 128 * 512);
    cvt(stream, in(S_W1), ws1, 384 * 384);
    cvt(stream, in(S_W2), ws2, 384 * 384);
    cvt(stream, in(S_W3), ws3, 384 * 384);
    for (int l = 0; l < 2; ++l) {
        int b = (l == 0) ? T0 : T1;
        cvt(stream, in(b + 0), wqkvL[l], 1152 * 384);
        cvt(stream, in(b + 2), woL[l], 384 * 384);
        cvt(stream, in(b + 6), w1L[l], 384 * 384);
        cvt(stream, in(b + 8), w2L[l], 384 * 384);
    }

    // ================= IPA =================
    // q, kv, q-points, kv-points projections
    run_fast<true,  true,  false, false>(stream, x16, wq,  q16,  in(I_BQ),  FR,
        256, 192, 384, 384, 384, 192, 0, 0, 0, 1, 1.f);
    run_fast<true,  true,  false, false>(stream, x16, wkv, kv16, in(I_BKV), FR,
        256, 384, 384, 384, 384, 384, 0, 0, 0, 1, 1.f);
    run_edge<true,  false>(stream, x16, wqp, qp_lin, in(I_BQP),
        256, 144, 384, 384, 384, 144, 0, 0, 0, 1, 1.f);
    run_edge<true,  false>(stream, x16, wkvp, kvp_lin, in(I_BKVP),
        256, 432, 384, 384, 384, 432, 0, 0, 0, 1, 1.f);
    k_make_pts<<<dim3((256 * 48 + 255) / 256), dim3(256), 0, stream>>>(
        qp_lin, in(IN_ROT), in(IN_TRANS), qpts, nullptr, 48, 4);
    k_make_pts<<<dim3((256 * 144 + 255) / 256), dim3(256), 0, stream>>>(
        kvp_lin, in(IN_ROT), in(IN_TRANS), kvpts, vpts16, 144, 12);

    // b_bias = z @ Wb^T ; pair_z = z @ Wdz^T (f16)
    run_edge<true,  false>(stream, z16, wb, bbias, in(I_BB),
        65536, 12, 128, 128, 128, 12, 0, 0, 0, 1, 1.f);
    run_fast<true,  true,  false, false>(stream, z16, wdz, pz16, in(I_BDZ), FR,
        65536, 32, 128, 128, 128, 32, 0, 0, 0, 1, 1.f);

    // attention logits: per head q @ k^T (K=16, zero-padded to 32)
    run_edge<true,  false>(stream, q16, kv16, qk32, F,
        256, 256, 16, 192, 384, 256, 16, 32, 65536, 12, AQK);
    k_ipa_softmax<<<dim3((12 * 256 * 32 + 127) / 128), dim3(128), 0, stream>>>(
        qk32, bbias, qpts, kvpts, in(IN_MASK), in(I_HW), a16);

    // o = a @ v  -> feats[:, 0:192] (f16, strided per head)
    run_edge<false, true >(stream, a16, kv16 + 16, feats, F,
        256, 16, 256, 256, 384, 960, 65536, 32, 16, 12, 1.f);
    // o_pt = a @ v_pts -> optraw (f32)
    run_edge<false, false>(stream, a16, vpts16, optraw, F,
        256, 24, 256, 256, 288, 288, 65536, 24, 24, 12, 1.f);
    k_opt_finish<<<dim3((256 * 96 + 255) / 256), dim3(256), 0, stream>>>(
        optraw, in(IN_ROT), in(IN_TRANS), feats);
    // o_pair: per-residue (batch=i) [12x256] @ [256x32] -> feats[:, 576:960]
    run_edge<false, true >(stream, a16, pz16, feats + 576, F,
        12, 32, 256, 65536, 32, 32, 256, 8192, 960, 256, 1.f);

    // IPA output projection + residual layernorm
    run_fast<true,  false, false, false>(stream, feats, wout, tmp32, in(I_BOUT), FR,
        256, 384, 960, 960, 960, 384, 0, 0, 0, 1, 1.f);
    k_ln<<<dim3((256 * 32 + 127) / 128), dim3(128), 0, stream>>>(
        in(IN_S), tmp32, in(IN_MASK), in(I_LNG), in(I_LNB), sbuf32, x16, 256, 384);

    // ================= Transformer x2 =================
    for (int l = 0; l < 2; ++l) {
        int b = (l == 0) ? T0 : T1;
        run_fast<true,  true,  false, false>(stream, x16, wqkvL[l], qkv16, in(b + 1), FR,
            256, 1152, 384, 384, 384, 1152, 0, 0, 0, 1, 1.f);
        // logits: q @ k^T per head (dh=96)
        run_fast<true,  false, false, false>(stream, qkv16, qkv16 + 384, qk32, F, FR,
            256, 256, 96, 1152, 1152, 256, 96, 96, 65536, 4, ATF);
        k_tf_softmax<<<dim3((4 * 256 * 32 + 127) / 128), dim3(128), 0, stream>>>(
            qk32, in(IN_MASK), a16);
        // o = att @ v
        run_fast<false, true,  false, false>(stream, a16, qkv16 + 768, oatt16, F, FR,
            256, 96, 256, 256, 1152, 384, 65536, 96, 96, 4, 1.f);
        run_fast<true,  false, false, false>(stream, oatt16, woL[l], tmp32, in(b + 3), FR,
            256, 384, 384, 384, 384, 384, 0, 0, 0, 1, 1.f);
        k_ln<<<dim3((256 * 32 + 127) / 128), dim3(128), 0, stream>>>(
            sbuf32, tmp32, nullptr, in(b + 4), in(b + 5), sbuf32, x16, 256, 384);
        // FFN
        run_fast<true,  true,  true,  false>(stream, x16, w1L[l], ffn16, in(b + 7), FR,
            256, 384, 384, 384, 384, 384, 0, 0, 0, 1, 1.f);
        run_fast<true,  false, false, false>(stream, ffn16, w2L[l], tmp32, in(b + 9), FR,
            256, 384, 384, 384, 384, 384, 0, 0, 0, 1, 1.f);
        k_ln<<<dim3((256 * 32 + 127) / 128), dim3(128), 0, stream>>>(
            sbuf32, tmp32, nullptr, in(b + 10), in(b + 11), sbuf32, x16, 256, 384);
    }

    // ================= SMT =================
    run_fast<true,  true,  true,  false>(stream, x16, ws1, t1_16, in(S_B1), FR,
        256, 384, 384, 384, 384, 384, 0, 0, 0, 1, 1.f);
    run_fast<true,  true,  true,  false>(stream, t1_16, ws2, t2_16, in(S_B2), FR,
        256, 384, 384, 384, 384, 384, 0, 0, 0, 1, 1.f);
    run_fast<true,  false, false, false>(stream, t2_16, ws3, tmp32, in(S_B3), FR,
        256, 384, 384, 384, 384, 384, 0, 0, 0, 1, 1.f);
    k_ln<<<dim3((256 * 32 + 127) / 128), dim3(128), 0, stream>>>(
        sbuf32, tmp32, nullptr, in(S_LNG), in(S_LNB), sbuf32, x16, 256, 384);

    // s = s * mask  -> d_out (node output) + f16 copy
    k_mask_s<<<dim3((256 * 384 + 255) / 256), dim3(256), 0, stream>>>(
        sbuf32, in(IN_MASK), out_s, x16);

    // rigid_update = s @ bb_W^T + bb_b -> d_out
    run_edge<true,  false>(stream, x16, wbbw, out_rig, in(I_BBB),
        256, 6, 384, 384, 384, 6, 0, 0, 0, 1, 1.f);

    // ================= Edge transition =================
    run_fast<true,  true,  false, false>(stream, x16, wi, n16, in(E_BI), FR,
        256, 192, 384, 384, 384, 192, 0, 0, 0, 1, 1.f);
    k_ecat<<<dim3((unsigned)(((long long)65536 * 512 + 255) / 256)), dim3(256), 0, stream>>>(
        z16, n16, ecat);
    // h1 = relu(e @ Wt1^T + bt1)
    run_fast<true,  true,  true,  false>(stream, ecat, wt1, h1buf, in(E_BT1), FR,
        65536, 512, 512, 512, 512, 512, 0, 0, 0, 1, 1.f);
    // t = relu(h1 @ Wt2^T + bt2) + e   (residual fused, written in-place over e)
    run_fast<true,  true,  true,  true >(stream, h1buf, wt2, ecat, in(E_BT2), ecat,
        65536, 512, 512, 512, 512, 512, 0, 0, 0, 1, 1.f);
    // edge = t @ Wf^T + bf  -> d_out, then in-place layernorm
    run_fast<true,  false, false, false>(stream, ecat, wf, out_e, in(E_BF), FR,
        65536, 128, 512, 512, 512, 128, 0, 0, 0, 1, 1.f);
    k_ln<<<dim3((65536 * 32 + 127) / 128), dim3(128), 0, stream>>>(
        out_e, nullptr, nullptr, in(E_LNG), in(E_LNB), out_e, nullptr, 65536, 128);
}
